// Generator_30528627540808
// MI455X (gfx1250) — matrix operand order repaired and checked by reading, not compiled, not run on hardware
//
#include <hip/hip_runtime.h>
#include <hip/hip_bf16.h>

typedef __attribute__((ext_vector_type(16))) _Float16 v16h;
typedef __attribute__((ext_vector_type(8)))  float    v8f;

// Problem constants
constexpr int cV  = 20000;   // vocab
constexpr int cH  = 400;     // hidden
constexpr int cHP = 416;     // hidden padded to 13*32 for WMMA K
constexpr int cKC = 13;      // K-chunks of 32
constexpr int cB  = 16;      // batch
constexpr int cK  = 30;      // slots
constexpr int cLS = 128;     // seq len
constexpr int cL  = 10;      // decode steps
constexpr int cROWS = cK * cB;          // 480
constexpr int cMT   = cROWS / 16;       // 30 M-tiles
constexpr int cNT_V = cV / 16;          // 1250 vocab N-tiles (even pairs)
constexpr int cNG   = 1200;             // gate width (3*H)
constexpr int cNGP  = 1216;             // gate width padded to even tile pairs
constexpr int cNT_G = cNGP / 16;        // 76 gate N-tiles (even pairs)
constexpr size_t GATE_OFF = (size_t)cK * cB * cL * cV;  // 96,000,000

// ---------------- prep kernels ----------------

__global__ void k_slot_emb(const float* __restrict__ swe, const int* __restrict__ dom,
                           const int* __restrict__ slt, float* __restrict__ se) {
  int id = blockIdx.x * 256 + threadIdx.x;
  if (id >= cK * cH) return;
  int k = id / cH, j = id % cH;
  se[id] = swe[(size_t)dom[k] * cH + j] + swe[(size_t)slt[k] * cH + j];
}

// Pack a row-major [nValid x 400] matrix (used as B = M^T, i.e. B[k][n] = M[n][k])
// into WMMA 16x16x32 f16 B-fragments: frag(nt,kc) -> 32 lanes x 16 halves,
// lane L: n = nt*16 + (L & 15), g = L>>4; half j: k = kc*32 + 8g + j (j<8) or kc*32 + 16 + 8g + (j-8) (j>=8).
// Rows n >= nValid and cols k >= 400 are zero-padded.
__global__ void k_pack_B(const float* __restrict__ M, _Float16* __restrict__ F,
                         int nTiles, int nValid) {
  size_t id = (size_t)blockIdx.x * 256 + threadIdx.x;
  size_t total = (size_t)nTiles * cKC * 512;
  if (id >= total) return;
  int j    = (int)(id & 15);
  int lane = (int)((id >> 4) & 31);
  size_t frag = id >> 9;
  int kc = (int)(frag % cKC);
  int nt = (int)(frag / cKC);
  int n  = nt * 16 + (lane & 15);
  int kk = kc * 32 + (lane >> 4) * 8 + j + ((j >= 8) ? 8 : 0);
  float v = (kk < cH && n < nValid) ? M[(size_t)n * cH + kk] : 0.0f;
  F[id] = (_Float16)v;
}

// Pack the f32 480x416 activation matrix into WMMA 16x16x32 f16 A-fragments:
// frag(mt,kc) -> 32 lanes x 16 halves; lane L: m = mt*16 + (L&15), g = L>>4;
// half j: k = kc*32 + 8g + j  (j<8)  or  kc*32 + 16 + 8g + (j-8)  (j>=8).
__global__ void k_pack_A(const float* __restrict__ A, _Float16* __restrict__ F) {
  int id = blockIdx.x * 256 + threadIdx.x;       // total = 30*13*512 = 199680
  if (id >= cMT * cKC * 512) return;
  int j    = id & 15;
  int lane = (id >> 4) & 31;
  int frag = id >> 9;
  int kc = frag % cKC;
  int mt = frag / cKC;
  int g = lane >> 4;
  int m = mt * 16 + (lane & 15);
  int k = kc * 32 + 8 * g + j + ((j >= 8) ? 8 : 0);
  F[id] = (_Float16)A[(size_t)m * cHP + k];      // A is zero-padded to 416 cols
}

// x[row] = slot_emb[k] (t==0) or E[target_batches[b,k,t-1]] (t>0), padded to 416
__global__ void k_gather_x(const float* __restrict__ E, const float* __restrict__ se,
                           const int* __restrict__ tgt, float* __restrict__ x, int t) {
  int id = blockIdx.x * 256 + threadIdx.x;
  if (id >= cROWS * cHP) return;
  int row = id / cHP, j = id % cHP;
  int k = row >> 4, b = row & 15;
  float v = 0.0f;
  if (j < cH) {
    if (t == 0) v = se[(size_t)k * cH + j];
    else {
      int tok = tgt[(size_t)b * cK * cL + (size_t)k * cL + (t - 1)];
      v = E[(size_t)tok * cH + j];
    }
  }
  x[id] = v;
}

// ---------------- WMMA GEMM: C[m, n] = sum_k A[m,k] * Bmat[n,k] ----------------
// Afrag: packed A-fragments (cMT x 13 x 512 halves)
// F:     packed B-fragments (nTiles x 13 x 512 halves); nTiles must be EVEN.
// Each wave computes TWO 16x16 D tiles (shared A fragment, two B streams),
// unconditionally -> no exec masking around WMMAs.
// C row address: Cbase + m*rowStride + n   (lets logits land directly in d_out)
__global__ void k_gemm(const _Float16* __restrict__ Afrag, const _Float16* __restrict__ F,
                       float* __restrict__ C, int nTiles, size_t rowStride) {
  int wave = threadIdx.x >> 5;
  int lane = threadIdx.x & 31;
  int nt0 = (blockIdx.x * 8 + wave) * 2;       // two N-tiles per wave
  int mt  = blockIdx.y;
  if (nt0 >= nTiles) return;                   // wave-uniform exit, EXEC stays all-1
  int g = lane >> 4, m16 = lane & 15;
  const _Float16* ap  = Afrag + ((size_t)mt * cKC) * 512 + (size_t)lane * 16;
  const _Float16* bp0 = F + ((size_t)nt0 * cKC) * 512 + (size_t)lane * 16;
  const _Float16* bp1 = bp0 + (size_t)cKC * 512;
  v8f acc0 = {}, acc1 = {};
#pragma unroll
  for (int kc = 0; kc < cKC; ++kc) {
    v16h a  = *(const v16h*)(ap  + (size_t)kc * 512);
    v16h b0 = *(const v16h*)(bp0 + (size_t)kc * 512);
    v16h b1 = *(const v16h*)(bp1 + (size_t)kc * 512);
    acc0 = __builtin_amdgcn_wmma_f32_16x16x32_f16(false, a, false, b0,
                                                  (short)0, acc0, false, false);
    acc1 = __builtin_amdgcn_wmma_f32_16x16x32_f16(false, a, false, b1,
                                                  (short)0, acc1, false, false);
  }
  // D layout: lane -> N = lane&15; VGPR r -> M = r + 8*(lane>>4)
  float* crow0 = C + (size_t)(mt * 16 + g * 8) * rowStride + (size_t)nt0 * 16 + m16;
  float* crow1 = crow0 + 16;
#pragma unroll
  for (int r = 0; r < 8; ++r) {
    crow0[(size_t)r * rowStride] = acc0[r];
    crow1[(size_t)r * rowStride] = acc1[r];
  }
}

// ---------------- GRU elementwise (in-place h update) ----------------
__global__ void k_gru(const float* __restrict__ gi, const float* __restrict__ gh,
                      const float* __restrict__ b_ih, const float* __restrict__ b_hh,
                      float* __restrict__ h) {
  int id = blockIdx.x * 256 + threadIdx.x;
  if (id >= cROWS * cHP) return;
  int row = id / cHP, j = id % cHP;
  float hn = 0.0f;
  if (j < cH) {
    const float* gir = gi + (size_t)row * cNGP;
    const float* ghr = gh + (size_t)row * cNGP;
    float ir = gir[j]       + b_ih[j],       hr_ = ghr[j]       + b_hh[j];
    float iz = gir[400 + j] + b_ih[400 + j], hz  = ghr[400 + j] + b_hh[400 + j];
    float in_= gir[800 + j] + b_ih[800 + j], hnn = ghr[800 + j] + b_hh[800 + j];
    float r = 1.0f / (1.0f + __expf(-(ir + hr_)));
    float z = 1.0f / (1.0f + __expf(-(iz + hz)));
    float n = tanhf(in_ + r * hnn);
    hn = (1.0f - z) * n + z * h[id];
  }
  h[id] = hn;
}

__global__ void k_h0(const float* __restrict__ cs, const float* __restrict__ cu,
                     float* __restrict__ h) {
  int id = blockIdx.x * 256 + threadIdx.x;
  if (id >= cROWS * cHP) return;
  int row = id / cHP, j = id % cHP;
  h[id] = (j < cH) ? cs[(size_t)row * cH + j] + cu[(size_t)row * cH + j] : 0.0f;
}

// ---------------- attention: scores->softmax->ctx (one block per (k,b) row) ----------------
__global__ void k_attend(const float* __restrict__ seqH, const float* __restrict__ musk,
                         const float* __restrict__ condH, const float* __restrict__ slotAtt,
                         int useSlot, float* __restrict__ ctx, float* __restrict__ prob) {
  __shared__ float cond_s[cH];
  __shared__ float pr[cLS];
  __shared__ float red[cLS];
  int row = blockIdx.x;
  int b = row & 15, k = row >> 4;
  int s = threadIdx.x;                       // blockDim = 128
  const float* cond = useSlot ? (slotAtt + (size_t)k * cH) : (condH + (size_t)row * cHP);
  for (int j = s; j < cH; j += cLS) cond_s[j] = cond[j];
  __syncthreads();
  const float* hrow = seqH + ((size_t)b * cLS + s) * cH;
  float dot = 0.0f;
  for (int j = 0; j < cH; ++j) dot += cond_s[j] * hrow[j];
  dot += musk[(size_t)b * cLS + s];
  red[s] = dot;
  __syncthreads();
  for (int off = 64; off > 0; off >>= 1) {
    if (s < off) red[s] = fmaxf(red[s], red[s + off]);
    __syncthreads();
  }
  float mx = red[0];
  __syncthreads();
  float e = __expf(dot - mx);
  red[s] = e;
  __syncthreads();
  for (int off = 64; off > 0; off >>= 1) {
    if (s < off) red[s] += red[s + off];
    __syncthreads();
  }
  float p = e / red[0];
  pr[s] = p;
  prob[(size_t)row * cLS + s] = p;
  __syncthreads();
  for (int d = s; d < cH; d += cLS) {
    float acc = 0.0f;
    const float* base = seqH + (size_t)b * cLS * cH + d;
    for (int t2 = 0; t2 < cLS; ++t2) acc += pr[t2] * base[(size_t)t2 * cH];
    ctx[(size_t)row * cH + d] = acc;
  }
}

// ---------------- per-row scalars: sw (ratio), alpha/beta (copy), gate @ t==0 ----------------
__global__ void k_small(const float* __restrict__ h, const float* __restrict__ cs,
                        const float* __restrict__ cu, const float* __restrict__ x,
                        const float* __restrict__ w_ratio, const float* __restrict__ b_ratio,
                        const float* __restrict__ w_copy, const float* __restrict__ b_copy,
                        const float* __restrict__ W_gate, const float* __restrict__ b_gate,
                        float* __restrict__ scal, float* __restrict__ gateOut, int writeGate) {
  __shared__ float red[6][256];
  int row = blockIdx.x, tid = threadIdx.x;
  const float* hr = h  + (size_t)row * cHP;
  const float* sr = cs + (size_t)row * cH;
  const float* ur = cu + (size_t)row * cH;
  const float* xr = x  + (size_t)row * cHP;
  float psw = 0, pcs = 0, pcu = 0, pg0 = 0, pg1 = 0, pg2 = 0;
  for (int i = tid; i < cH; i += 256) {
    float hv = hr[i], sv = sr[i], uv = ur[i], xv = xr[i];
    // pgen = [h, ctx_s, ctx_u, x] . w_ratio
    psw += hv * w_ratio[i] + sv * w_ratio[400 + i] + uv * w_ratio[800 + i] + xv * w_ratio[1200 + i];
    // c_s = [h, ctx_s, x] . w_copy ; c_u = [h, ctx_u, x] . w_copy
    pcs += hv * w_copy[i] + sv * w_copy[400 + i] + xv * w_copy[800 + i];
    pcu += hv * w_copy[i] + uv * w_copy[400 + i] + xv * w_copy[800 + i];
    // gate = [ctx_s; ctx_u] @ W_gate(800x3)
    pg0 += sv * W_gate[(size_t)i * 3 + 0] + uv * W_gate[(size_t)(400 + i) * 3 + 0];
    pg1 += sv * W_gate[(size_t)i * 3 + 1] + uv * W_gate[(size_t)(400 + i) * 3 + 1];
    pg2 += sv * W_gate[(size_t)i * 3 + 2] + uv * W_gate[(size_t)(400 + i) * 3 + 2];
  }
  red[0][tid] = psw; red[1][tid] = pcs; red[2][tid] = pcu;
  red[3][tid] = pg0; red[4][tid] = pg1; red[5][tid] = pg2;
  __syncthreads();
  for (int off = 128; off > 0; off >>= 1) {
    if (tid < off)
      for (int q = 0; q < 6; ++q) red[q][tid] += red[q][tid + off];
    __syncthreads();
  }
  if (tid == 0) {
    float sw = 1.0f / (1.0f + __expf(-(red[0][0] + b_ratio[0])));
    float csc = red[1][0] + b_copy[0];
    float cuc = red[2][0] + b_copy[0];
    float m = fmaxf(csc, cuc);
    float es = __expf(csc - m), eu = __expf(cuc - m);
    float inv = 1.0f / (es + eu);
    scal[row * 4 + 0] = sw;
    scal[row * 4 + 1] = es * inv;   // alpha
    scal[row * 4 + 2] = eu * inv;   // beta
    if (writeGate) {
      gateOut[row * 3 + 0] = red[3][0] + b_gate[0];
      gateOut[row * 3 + 1] = red[4][0] + b_gate[1];
      gateOut[row * 3 + 2] = red[5][0] + b_gate[2];
    }
  }
}

// ---------------- in-place softmax over logits row in d_out, scale by sw ----------------
__global__ void k_softmax_write(float* __restrict__ out, const float* __restrict__ scal, int t) {
  __shared__ float red[256];
  int row = blockIdx.x, tid = threadIdx.x;
  float* orow = out + ((size_t)row * cL + t) * cV;
  float mx = -3.4e38f;
  for (int v = tid; v < cV; v += 256) mx = fmaxf(mx, orow[v]);
  red[tid] = mx;
  __syncthreads();
  for (int off = 128; off > 0; off >>= 1) {
    if (tid < off) red[tid] = fmaxf(red[tid], red[tid + off]);
    __syncthreads();
  }
  mx = red[0];
  __syncthreads();
  float sum = 0.0f;
  for (int v = tid; v < cV; v += 256) sum += __expf(orow[v] - mx);
  red[tid] = sum;
  __syncthreads();
  for (int off = 128; off > 0; off >>= 1) {
    if (tid < off) red[tid] += red[tid + off];
    __syncthreads();
  }
  float scale = scal[row * 4 + 0] / red[0];   // sw / denom
  for (int v = tid; v < cV; v += 256) orow[v] = scale * __expf(orow[v] - mx);
}

// ---------------- sparse pointer scatter: += (1-sw)*{alpha,beta}*prob at story ids ----------------
__global__ void k_scatter(const float* __restrict__ probS, const float* __restrict__ probU,
                          const int* __restrict__ storyS, const int* __restrict__ storyU,
                          const float* __restrict__ scal, float* __restrict__ out, int t) {
  int row = blockIdx.x, tid = threadIdx.x;
  int b = row & 15;
  float sw = scal[row * 4 + 0], alpha = scal[row * 4 + 1], beta = scal[row * 4 + 2];
  float* orow = out + ((size_t)row * cL + t) * cV;
  if (tid < cLS) {
    int idx = storyS[(size_t)b * cLS + tid];
    atomicAdd(&orow[idx], (1.0f - sw) * alpha * probS[(size_t)row * cLS + tid]);
  } else {
    int s2 = tid - cLS;
    int idx = storyU[(size_t)b * cLS + s2];
    atomicAdd(&orow[idx], (1.0f - sw) * beta * probU[(size_t)row * cLS + s2]);
  }
}

// ---------------- host launcher ----------------
extern "C" void kernel_launch(void* const* d_in, const int* in_sizes, int n_in,
                              void* d_out, int out_size, void* d_ws, size_t ws_size,
                              hipStream_t stream) {
  const float* sys_H     = (const float*)d_in[0];
  const float* user_H    = (const float*)d_in[1];
  const float* sys_musk  = (const float*)d_in[2];
  const float* user_musk = (const float*)d_in[3];
  const int*   story_sys = (const int*)d_in[4];
  const int*   story_user= (const int*)d_in[5];
  const int*   targets   = (const int*)d_in[6];
  const int*   domain_idx= (const int*)d_in[7];
  const int*   slot_idx  = (const int*)d_in[8];
  const float* E         = (const float*)d_in[9];
  const float* W_ih      = (const float*)d_in[10];
  const float* W_hh      = (const float*)d_in[11];
  const float* b_ih      = (const float*)d_in[12];
  const float* b_hh      = (const float*)d_in[13];
  const float* w_copy    = (const float*)d_in[14];
  const float* b_copy    = (const float*)d_in[15];
  const float* w_ratio   = (const float*)d_in[16];
  const float* b_ratio   = (const float*)d_in[17];
  const float* W_gate    = (const float*)d_in[18];
  const float* b_gate    = (const float*)d_in[19];
  const float* slot_att  = (const float*)d_in[20];
  const float* slot_wemb = (const float*)d_in[21];
  float* out = (float*)d_out;

  // carve workspace (~28 MB)
  char* w = (char*)d_ws;
  auto carve = [&](size_t bytes) { char* p = w; w += (bytes + 255) & ~(size_t)255; return (void*)p; };
  _Float16* Efrag  = (_Float16*)carve((size_t)cNT_V * cKC * 512 * 2);
  _Float16* WihF   = (_Float16*)carve((size_t)cNT_G * cKC * 512 * 2);
  _Float16* WhhF   = (_Float16*)carve((size_t)cNT_G * cKC * 512 * 2);
  _Float16* AfragH = (_Float16*)carve((size_t)cMT * cKC * 512 * 2);
  _Float16* AfragX = (_Float16*)carve((size_t)cMT * cKC * 512 * 2);
  float* slotEmb  = (float*)carve((size_t)cK * cH * 4);
  float* hBuf     = (float*)carve((size_t)cROWS * cHP * 4);
  float* xBuf     = (float*)carve((size_t)cROWS * cHP * 4);
  float* giBuf    = (float*)carve((size_t)cROWS * cNGP * 4);
  float* ghBuf    = (float*)carve((size_t)cROWS * cNGP * 4);
  float* ctxS     = (float*)carve((size_t)cROWS * cH * 4);
  float* ctxU     = (float*)carve((size_t)cROWS * cH * 4);
  float* probS    = (float*)carve((size_t)cROWS * cLS * 4);
  float* probU    = (float*)carve((size_t)cROWS * cLS * 4);
  float* scal     = (float*)carve((size_t)cROWS * 4 * 4);

  // one-time prep: slot embeddings + f16 B-fragment packing of E, W_ih, W_hh
  k_slot_emb<<<(cK * cH + 255) / 256, 256, 0, stream>>>(slot_wemb, domain_idx, slot_idx, slotEmb);
  k_pack_B<<<(int)(((size_t)cNT_V * cKC * 512 + 255) / 256), 256, 0, stream>>>(E, Efrag, cNT_V, cV);
  k_pack_B<<<(int)(((size_t)cNT_G * cKC * 512 + 255) / 256), 256, 0, stream>>>(W_ih, WihF, cNT_G, cNG);
  k_pack_B<<<(int)(((size_t)cNT_G * cKC * 512 + 255) / 256), 256, 0, stream>>>(W_hh, WhhF, cNT_G, cNG);

  // h0 = attend(sys, slot_att) + attend(user, slot_att)
  k_attend<<<cROWS, cLS, 0, stream>>>(sys_H,  sys_musk,  nullptr, slot_att, 1, ctxS, probS);
  k_attend<<<cROWS, cLS, 0, stream>>>(user_H, user_musk, nullptr, slot_att, 1, ctxU, probU);
  k_h0<<<(cROWS * cHP + 255) / 256, 256, 0, stream>>>(ctxS, ctxU, hBuf);

  float* gateOut = out + GATE_OFF;
  const int grid_e = cROWS * cHP / 256;               // 780
  const int grid_a = cMT * cKC * 512 / 256;           // 780
  const int gx_g   = (cNT_G / 2 + 7) / 8;             // 5  blocks for N=1216
  const int gx_v   = (cNT_V / 2 + 7) / 8;             // 79 blocks for N=20000

  k_pack_A<<<grid_a, 256, 0, stream>>>(hBuf, AfragH);  // h0 fragments

  for (int t = 0; t < cL; ++t) {
    k_gather_x<<<grid_e, 256, 0, stream>>>(E, slotEmb, targets, xBuf, t);
    k_pack_A<<<grid_a, 256, 0, stream>>>(xBuf, AfragX);
    // gi = x @ W_ih^T ; gh = h_prev @ W_hh^T   (WMMA f16, N padded to 1216)
    k_gemm<<<dim3(gx_g, cMT), 256, 0, stream>>>(AfragX, WihF, giBuf, cNT_G, cNGP);
    k_gemm<<<dim3(gx_g, cMT), 256, 0, stream>>>(AfragH, WhhF, ghBuf, cNT_G, cNGP);
    k_gru<<<grid_e, 256, 0, stream>>>(giBuf, ghBuf, b_ih, b_hh, hBuf);
    k_pack_A<<<grid_a, 256, 0, stream>>>(hBuf, AfragH);   // h_new fragments (logits now, gh next step)
    k_attend<<<cROWS, cLS, 0, stream>>>(sys_H,  sys_musk,  hBuf, nullptr, 0, ctxS, probS);
    k_attend<<<cROWS, cLS, 0, stream>>>(user_H, user_musk, hBuf, nullptr, 0, ctxU, probU);
    k_small<<<cROWS, 256, 0, stream>>>(hBuf, ctxS, ctxU, xBuf, w_ratio, b_ratio,
                                       w_copy, b_copy, W_gate, b_gate, scal, gateOut,
                                       (t == 0) ? 1 : 0);
    // logits = h_new @ E^T, written straight into d_out row (row stride cL*cV)
    k_gemm<<<dim3(gx_v, cMT), 256, 0, stream>>>(AfragH, Efrag, out + (size_t)t * cV,
                                                cNT_V, (size_t)cL * cV);
    k_softmax_write<<<cROWS, 256, 0, stream>>>(out, scal, t);
    k_scatter<<<cROWS, 256, 0, stream>>>(probS, probU, story_sys, story_user, scal, out, t);
  }
}
